// BCESleepLoss_74414603370694
// MI455X (gfx1250) — matrix operand order repaired and checked by reading, not compiled, not run on hardware
//
#include <hip/hip_runtime.h>
#include <math.h>

// Problem constants (from reference)
#define NB   32
#define LSIG 8192
#define CCH  3
#define LAMBDA1 1.0f
#define LAMBDA2 1.0e-4f

// Cross-correlation kernel configuration
#define LAGS_PER_WG 2048      // each workgroup covers 2048 lags
#define TILE_LAGS 256         // one 16x16 WMMA tile = 256 lags
#define CH 2112               // i0 chunk length (multiple of 32)
#define KSTEPS (CH / 32)      // K=32 contraction steps per chunk
#define NCHUNK 4              // 4*2112 = 8448 >= 8416+32 needed for full coverage
#define HA 256                // left halo for the A (Toeplitz) signal
#define ALDSE (CH + 272)      // staged a elements (with halos)
#define BLDSE (CH + 1824)     // staged b elements (2048-lag span + K + N)
#define APK (ALDSE / 2)       // packed bf16 pairs for a
#define BPK (BLDSE / 2)       // packed bf16 pairs for b (per phase)

typedef unsigned int u32;
typedef u32    v4u  __attribute__((ext_vector_type(4)));
typedef u32    v8u  __attribute__((ext_vector_type(8)));
typedef float  v8f  __attribute__((ext_vector_type(8)));
typedef __bf16 v16bf __attribute__((ext_vector_type(16)));
// Under-aligned vector view: B pair-index is only dword-aligned, so load the
// whole 8-dword operand as one SSA value (backend splits into legal DS ops,
// destination stays one contiguous VGPR tuple -> no mov-assembly, fewer
// WMMA hazard v_nops).
typedef v8u v8u_u __attribute__((aligned(4)));

// Split x into hi/lo bf16 parts and return both as raw 16-bit patterns.
static __device__ inline void split_bf16(float x, unsigned short& hi, unsigned short& lo) {
  __bf16 h = (__bf16)x;
  float  r = x - (float)h;
  __bf16 l = (__bf16)r;
  hi = __builtin_bit_cast(unsigned short, h);
  lo = __builtin_bit_cast(unsigned short, l);
}
static __device__ inline u32 pack2(unsigned short a, unsigned short b) {
  return (u32)a | ((u32)b << 16);
}

// ---------------------------------------------------------------------------
// Kernel 1: weighted BCE-with-logits partial sums (elementwise, memory-bound)
// ---------------------------------------------------------------------------
__global__ void bce_partial_kernel(const float* __restrict__ pred,
                                   const float* __restrict__ targ,
                                   const float* __restrict__ wgt,
                                   float* __restrict__ part) {
  __shared__ float red[256];
  const int total = NB * LSIG * CCH;
  const float w0 = wgt[0], w1 = wgt[1], w2 = wgt[2];
  float acc = 0.0f;
  for (int idx = blockIdx.x * blockDim.x + threadIdx.x; idx < total;
       idx += gridDim.x * blockDim.x) {
    float x = pred[idx];
    float t = targ[idx];
    int   c = idx % 3;
    float w = (c == 0) ? w0 : ((c == 1) ? w1 : w2);
    float v = fmaxf(x, 0.0f) - x * t + log1pf(expf(-fabsf(x)));
    acc += v * w;
  }
  red[threadIdx.x] = acc;
  __syncthreads();
  for (int s = 128; s > 0; s >>= 1) {
    if (threadIdx.x < s) red[threadIdx.x] += red[threadIdx.x + s];
    __syncthreads();
  }
  if (threadIdx.x == 0) part[blockIdx.x] = red[0];
}

// ---------------------------------------------------------------------------
// Kernel 2: cross-correlation energy via bf16 hi/lo-split WMMA Toeplitz tiles.
//   c[m] = sum_i s1[i]*s2[i+m],  m in [-8191, 8191]
//   A[r,k] = s1[i0+k-16r], B[k,n] = s2[i0+k+m0+n]
//     => D[r,n] accumulates c[m0 + 16r + n] over the i contraction.
//   Each f32 value is split a = ah + al (bf16); c accumulated as
//   ah*bh + ah*bl + al*bh in f32 WMMA accumulators (~1e-5 rel error).
//   Each wave owns one 256-lag tile; 8 waves/WG; 256 WGs = 32 batches x 8.
// ---------------------------------------------------------------------------
__global__ void corr_partial_kernel(const float* __restrict__ pred,
                                    float* __restrict__ part) {
  // Packed-pair bf16 staging: one dword = two adjacent bf16 elements.
  __shared__ __align__(16) u32 apk_hi[APK];
  __shared__ __align__(16) u32 apk_lo[APK];
  __shared__ __align__(16) u32 bpk_hi[2 * BPK];  // [0,BPK) even phase, [BPK,2BPK) odd
  __shared__ __align__(16) u32 bpk_lo[2 * BPK];
  __shared__ float red[256];
  __shared__ float s_norm;

  const int wg  = blockIdx.x;        // 0..255
  const int b   = wg >> 3;           // batch
  const int g   = wg & 7;            // lag group
  const int mg  = -LSIG + g * LAGS_PER_WG;   // first lag of this workgroup
  const int tid = threadIdx.x;

  const float* s1 = pred + (size_t)b * LSIG * CCH + 1;  // channel 1, stride 3
  const float* s2 = pred + (size_t)b * LSIG * CCH + 2;  // channel 2, stride 3

  // ---- per-batch norms in exact f32: sqrt(sum(s1^2) * sum(s2^2)) ----
  float n1 = 0.0f, n2 = 0.0f;
  for (int i = tid; i < LSIG; i += 256) {
    float x = s1[i * 3];
    float y = s2[i * 3];
    n1 += x * x;
    n2 += y * y;
  }
  red[tid] = n1;
  __syncthreads();
  for (int s = 128; s > 0; s >>= 1) {
    if (tid < s) red[tid] += red[tid + s];
    __syncthreads();
  }
  float sum1 = red[0];
  __syncthreads();
  red[tid] = n2;
  __syncthreads();
  for (int s = 128; s > 0; s >>= 1) {
    if (tid < s) red[tid] += red[tid + s];
    __syncthreads();
  }
  if (tid == 0) s_norm = sqrtf(sum1 * red[0]);

  // ---- WMMA lane indexing (16-bit operand layouts, wave32) ----
  // A 16x32 bf16: lane l -> M=l%16; VGPR v: K = 2v + 16*(l>=16)  (k order used here)
  //                                 (pairs K,K+1; same k order as B)
  // B 32x16 bf16: lane l -> N=l%16; VGPR v: K = 2v + 16*(l>=16) as for A (this code's order)       (pairs K,K+1)
  // D 16x16 f32 : VGPR j, lane l -> M = j + 8*(l>=16), N = l%16
  const int lane = tid & 31;
  const int wv   = tid >> 5;          // wave id: tile m0 = mg + 256*wv
  const int half = lane >> 4;         // 0 or 1
  const int lp   = lane & 15;         // M (for A) / N (for B)
  const int bphase = (lp & 1) * BPK;  // odd N -> odd-phase packed array

  v8f acc = {0.f, 0.f, 0.f, 0.f, 0.f, 0.f, 0.f, 0.f};

  for (int ch = 0; ch < NCHUNK; ++ch) {
    const int ibase = ch * CH;
    __syncthreads();
    // Stage a: pair j packs elements (2j, 2j+1) of [ibase-HA, ...), hi & lo.
    for (int j = tid; j < APK; j += 256) {
      int g0 = ibase - HA + 2 * j;
      float x0 = (g0 >= 0 && g0 < LSIG) ? s1[g0 * 3] : 0.0f;
      float x1 = (g0 + 1 >= 0 && g0 + 1 < LSIG) ? s1[(g0 + 1) * 3] : 0.0f;
      unsigned short h0, l0, h1, l1;
      split_bf16(x0, h0, l0);
      split_bf16(x1, h1, l1);
      apk_hi[j] = pack2(h0, h1);
      apk_lo[j] = pack2(l0, l1);
    }
    // Stage b: even phase packs (2j,2j+1), odd phase packs (2j+1,2j+2),
    // elements relative to global index (ibase + mg).
    for (int j = tid; j < BPK; j += 256) {
      int g0 = ibase + mg + 2 * j;
      float x0 = (g0 >= 0 && g0 < LSIG) ? s2[g0 * 3] : 0.0f;
      float x1 = (g0 + 1 >= 0 && g0 + 1 < LSIG) ? s2[(g0 + 1) * 3] : 0.0f;
      float x2 = (g0 + 2 >= 0 && g0 + 2 < LSIG) ? s2[(g0 + 2) * 3] : 0.0f;
      unsigned short h0, l0, h1, l1, h2, l2;
      split_bf16(x0, h0, l0);
      split_bf16(x1, h1, l1);
      split_bf16(x2, h2, l2);
      bpk_hi[j]       = pack2(h0, h1);
      bpk_hi[BPK + j] = pack2(h1, h2);
      bpk_lo[j]       = pack2(l0, l1);
      bpk_lo[BPK + j] = pack2(l1, l2);
    }
    __syncthreads();

    // Pair-index bases (advance by 16 pairs per K=32 step).
    // apj = 16*s + 128 - 8*lp + 8*half  (always a multiple of 4 -> b128 loads)
    int apj = 128 - 8 * lp + 8 * half;
    // bpj = 16*s + 8*half + 128*wv + (lp>>1)  (+ phase array offset)
    int bpj = 8 * half + 128 * wv + (lp >> 1) + bphase;

#pragma unroll 2
    for (int s = 0; s < KSTEPS; ++s) {
      // A operands: two aligned 16B reads each (VGPR groups 0-3 and 4-7).
      v4u ah0 = *(const v4u*)(apk_hi + apj);
      v4u ah1 = *(const v4u*)(apk_hi + apj + 4);
      v4u al0 = *(const v4u*)(apk_lo + apj);
      v4u al1 = *(const v4u*)(apk_lo + apj + 4);
      // B operands: one under-aligned 8-dword vector load each.
      v8u bhu = *(const v8u_u*)(bpk_hi + bpj);
      v8u blu = *(const v8u_u*)(bpk_lo + bpj);

      v8u ahu, alu;
      ahu[0] = ah0[0]; ahu[1] = ah0[1]; ahu[2] = ah0[2]; ahu[3] = ah0[3];
      ahu[4] = ah1[0]; ahu[5] = ah1[1]; ahu[6] = ah1[2]; ahu[7] = ah1[3];
      alu[0] = al0[0]; alu[1] = al0[1]; alu[2] = al0[2]; alu[3] = al0[3];
      alu[4] = al1[0]; alu[5] = al1[1]; alu[6] = al1[2]; alu[7] = al1[3];

      v16bf Ah = __builtin_bit_cast(v16bf, ahu);
      v16bf Al = __builtin_bit_cast(v16bf, alu);
      v16bf Bh = __builtin_bit_cast(v16bf, bhu);
      v16bf Bl = __builtin_bit_cast(v16bf, blu);

      // c += ah*bh + ah*bl + al*bh   (al*bl dropped: ~2^-18 relative)
      acc = __builtin_amdgcn_wmma_f32_16x16x32_bf16(
          false, Ah, false, Bh, (short)0, acc, false, false);
      acc = __builtin_amdgcn_wmma_f32_16x16x32_bf16(
          false, Ah, false, Bl, (short)0, acc, false, false);
      acc = __builtin_amdgcn_wmma_f32_16x16x32_bf16(
          false, Al, false, Bh, (short)0, acc, false, false);

      apj += 16;
      bpj += 16;
    }
  }
  __syncthreads();

  // ---- sum of squared correlations over this workgroup's 2048 lags ----
  float p = 0.0f;
#pragma unroll
  for (int j = 0; j < 8; ++j) p += acc[j] * acc[j];
  red[tid] = p;
  __syncthreads();
  for (int s = 128; s > 0; s >>= 1) {
    if (tid < s) red[tid] += red[tid + s];
    __syncthreads();
  }
  if (tid == 0) part[wg] = red[0] / s_norm;
}

// ---------------------------------------------------------------------------
// Kernel 3: deterministic finalize (single block)
// ---------------------------------------------------------------------------
__global__ void finalize_kernel(const float* __restrict__ bce_part,
                                const float* __restrict__ corr_part,
                                float* __restrict__ out) {
  __shared__ float red[256];
  const int tid = threadIdx.x;

  red[tid] = bce_part[tid];
  __syncthreads();
  for (int s = 128; s > 0; s >>= 1) {
    if (tid < s) red[tid] += red[tid + s];
    __syncthreads();
  }
  float bce_sum = red[0];
  __syncthreads();

  red[tid] = corr_part[tid];
  __syncthreads();
  for (int s = 128; s > 0; s >>= 1) {
    if (tid < s) red[tid] += red[tid + s];
    __syncthreads();
  }
  if (tid == 0) {
    float bce_mean = bce_sum / (float)(NB * LSIG * CCH);
    out[0] = LAMBDA1 * bce_mean + LAMBDA2 * red[0];
  }
}

// ---------------------------------------------------------------------------
extern "C" void kernel_launch(void* const* d_in, const int* in_sizes, int n_in,
                              void* d_out, int out_size, void* d_ws, size_t ws_size,
                              hipStream_t stream) {
  const float* pred = (const float*)d_in[0];
  const float* targ = (const float*)d_in[1];
  const float* wgt  = (const float*)d_in[2];
  float* out = (float*)d_out;
  float* ws  = (float*)d_ws;

  float* bce_part  = ws;        // 256 floats
  float* corr_part = ws + 256;  // 256 floats

  bce_partial_kernel<<<256, 256, 0, stream>>>(pred, targ, wgt, bce_part);
  corr_partial_kernel<<<256, 256, 0, stream>>>(pred, corr_part);
  finalize_kernel<<<1, 256, 0, stream>>>(bce_part, corr_part, out);
}
